// FocalNet_2D_24300924961179
// MI455X (gfx1250) — compile-verified, matrix operand order checked by reading, not run on hardware
//
#include <hip/hip_runtime.h>
#include <math.h>

// ---------------------------------------------------------------------------
// Types for CDNA5 WMMA (wave32)
// ---------------------------------------------------------------------------
typedef __bf16 bf16;
typedef __attribute__((ext_vector_type(16))) __bf16 v16bf;
typedef __attribute__((ext_vector_type(8)))  float  v8f;

#define TOKENS   65536L      // 16 * 64 * 64
#define CH       192
#define HDIM     768
#define NFPAD    400         // 388 padded to 16-multiple
#define SPATIAL  4096L       // 64*64 per batch image

__device__ __forceinline__ float gelu_f(float v) {
    return 0.5f * v * (1.0f + erff(v * 0.70710678118654752f));
}

// ---------------------------------------------------------------------------
// Multi-tile wave-level WMMA core.
//   A: row-major bf16 [M x K] (rows 16B aligned; lda multiple of 8)
//   Bp: fragment-packed weights, layout [N/16][K/32][lane:32][16 bf16]
// Each wave computes one 16 x (NTW*16) strip: A fragment loaded once per
// K-step (2 x b128), each B fragment is 32 contiguous bytes (2 x b128).
// ---------------------------------------------------------------------------
template<int NTW>
__device__ __forceinline__ void wmma_strip(const bf16* __restrict__ A, int lda,
                                           const bf16* __restrict__ Bp, int K,
                                           long m0, int nt0, int lane,
                                           v8f acc[NTW]) {
    const int half = lane >> 4;
    const int l15  = lane & 15;
    const bf16* __restrict__ arow = A + (m0 + l15) * (long)lda;
    const int KT = K >> 5;
#pragma unroll
    for (int t = 0; t < NTW; ++t) acc[t] = v8f{};
    for (int kt = 0; kt < KT; ++kt) {
        const int k0 = kt << 5;
        v16bf av;
        *(uint4*)&av       = *(const uint4*)(arow + k0 + 8 * half);
        *((uint4*)&av + 1) = *(const uint4*)(arow + k0 + 16 + 8 * half);
#pragma unroll
        for (int t = 0; t < NTW; ++t) {
            const bf16* __restrict__ bp =
                Bp + ((long)((nt0 + t) * KT + kt) * 32 + lane) * 16;
            v16bf bv;
            *(uint4*)&bv       = *(const uint4*)bp;
            *((uint4*)&bv + 1) = *(const uint4*)(bp + 8);
            acc[t] = __builtin_amdgcn_wmma_f32_16x16x32_bf16(
                false, av, false, bv, (short)0, acc[t], false, false);
        }
    }
}

// ---------------------------------------------------------------------------
// Pack fp32 weights [K x n_src] into bf16 WMMA-B fragment order
// dst[(nt*KT + kt)*512 + lane*16 + e]; zero-fill padded columns.
//   half = lane>>4; n = nt*16 + (lane&15)
//   koff = e<8 ? 8*half + e : 8*half + e + 8 ; k = kt*32 + koff
// ---------------------------------------------------------------------------
__global__ __launch_bounds__(256)
void wpack_kernel(const float* __restrict__ src, bf16* __restrict__ dst,
                  int K, int n_src, int n_pad) {
    const int KT = K >> 5;
    long i = (long)blockIdx.x * blockDim.x + threadIdx.x;
    long tot = (long)(n_pad >> 4) * KT * 512;
    if (i >= tot) return;
    const int  e    = (int)(i & 15);
    const int  lane = (int)((i >> 4) & 31);
    const long blk  = i >> 9;                 // nt*KT + kt
    const int  kt   = (int)(blk % KT);
    const int  nt   = (int)(blk / KT);
    const int  h    = lane >> 4;
    const int  n    = nt * 16 + (lane & 15);
    const int  koff = (e < 8) ? (8 * h + e) : (8 * h + e + 8);
    const int  k    = kt * 32 + koff;
    dst[i] = (bf16)((n < n_src) ? src[(long)k * n_src + n] : 0.0f);
}

// ---------------------------------------------------------------------------
// LayerNorm (C=192) + convert to bf16.  One wave per token, 6 ch/lane.
// ---------------------------------------------------------------------------
__global__ __launch_bounds__(256)
void ln_bf16_kernel(const float* __restrict__ x, const float* __restrict__ w,
                    const float* __restrict__ b, bf16* __restrict__ y, long ntok) {
    const int wave = threadIdx.x >> 5;
    const int lane = threadIdx.x & 31;
    const long tok = (long)blockIdx.x * (blockDim.x >> 5) + wave;
    if (tok >= ntok) return;
    const float* __restrict__ row = x + tok * CH;
    float v[6], s = 0.f, s2 = 0.f;
#pragma unroll
    for (int i = 0; i < 6; ++i) {
        v[i] = row[lane + 32 * i];
        s  += v[i];
        s2 += v[i] * v[i];
    }
#pragma unroll
    for (int m = 16; m >= 1; m >>= 1) {
        s  += __shfl_xor(s,  m, 32);
        s2 += __shfl_xor(s2, m, 32);
    }
    const float mean = s * (1.0f / CH);
    const float var  = s2 * (1.0f / CH) - mean * mean;
    const float rs   = rsqrtf(var + 1e-5f);
    bf16* __restrict__ orow = y + tok * CH;
#pragma unroll
    for (int i = 0; i < 6; ++i) {
        const int c = lane + 32 * i;
        orow[c] = (bf16)((v[i] - mean) * rs * w[c] + b[c]);
    }
}

// ---------------------------------------------------------------------------
// GEMM1: t = y @ f_w + f_b  (N=400 pad, 5 tiles/wave) -> split q / ctx / gates
// ---------------------------------------------------------------------------
__global__ __launch_bounds__(256)
void gemm1_kernel(const bf16* __restrict__ A, const bf16* __restrict__ Bp,
                  const float* __restrict__ fb, float* __restrict__ q,
                  float* __restrict__ ctx, float* __restrict__ gates, int mtiles) {
    const int NTW = 5, GROUPS = 5;            // 25 n-tiles
    const int wave = threadIdx.x >> 5, lane = threadIdx.x & 31;
    const int grp = blockIdx.x * 8 + wave;
    if (grp >= mtiles * GROUPS) return;       // wave-uniform
    const long m0  = (long)(grp / GROUPS) * 16;
    const int  nt0 = (grp % GROUPS) * NTW;
    v8f acc[NTW];
    wmma_strip<NTW>(A, CH, Bp, CH, m0, nt0, lane, acc);
    const int half = lane >> 4, l15 = lane & 15;
#pragma unroll
    for (int t = 0; t < NTW; ++t) {
        const int j = (nt0 + t) * 16 + l15;
#pragma unroll
        for (int v = 0; v < 8; ++v) {
            const long row = m0 + v + 8 * half;
            if (j < 388) {
                const float val = acc[t][v] + fb[j];
                if      (j < CH)   q[row * CH + j]             = val;
                else if (j < 2*CH) ctx[row * CH + (j - CH)]    = val;
                else               gates[row * 4 + (j - 2*CH)] = val;
            }
        }
    }
}

// ---------------------------------------------------------------------------
// GEMM2: modulator = ctx_all_bf @ h_w + h_b ; fused qmod = q*modulator -> bf16
// ---------------------------------------------------------------------------
__global__ __launch_bounds__(256)
void gemm2_kernel(const bf16* __restrict__ A, const bf16* __restrict__ Bp,
                  const float* __restrict__ hb, const float* __restrict__ q,
                  bf16* __restrict__ qmod, int mtiles) {
    const int NTW = 4, GROUPS = 3;            // 12 n-tiles
    const int wave = threadIdx.x >> 5, lane = threadIdx.x & 31;
    const int grp = blockIdx.x * 8 + wave;
    if (grp >= mtiles * GROUPS) return;
    const long m0  = (long)(grp / GROUPS) * 16;
    const int  nt0 = (grp % GROUPS) * NTW;
    v8f acc[NTW];
    wmma_strip<NTW>(A, CH, Bp, CH, m0, nt0, lane, acc);
    const int half = lane >> 4, l15 = lane & 15;
#pragma unroll
    for (int t = 0; t < NTW; ++t) {
        const int j = (nt0 + t) * 16 + l15;
#pragma unroll
        for (int v = 0; v < 8; ++v) {
            const long row = m0 + v + 8 * half;
            qmod[row * CH + j] = (bf16)(q[row * CH + j] * (acc[t][v] + hb[j]));
        }
    }
}

// ---------------------------------------------------------------------------
// GEMM3: x1 = x + qmod_bf @ proj_w + proj_b
// ---------------------------------------------------------------------------
__global__ __launch_bounds__(256)
void gemm3_kernel(const bf16* __restrict__ A, const bf16* __restrict__ Bp,
                  const float* __restrict__ pb, const float* __restrict__ xin,
                  float* __restrict__ x1, int mtiles) {
    const int NTW = 4, GROUPS = 3;
    const int wave = threadIdx.x >> 5, lane = threadIdx.x & 31;
    const int grp = blockIdx.x * 8 + wave;
    if (grp >= mtiles * GROUPS) return;
    const long m0  = (long)(grp / GROUPS) * 16;
    const int  nt0 = (grp % GROUPS) * NTW;
    v8f acc[NTW];
    wmma_strip<NTW>(A, CH, Bp, CH, m0, nt0, lane, acc);
    const int half = lane >> 4, l15 = lane & 15;
#pragma unroll
    for (int t = 0; t < NTW; ++t) {
        const int j = (nt0 + t) * 16 + l15;
#pragma unroll
        for (int v = 0; v < 8; ++v) {
            const long row = m0 + v + 8 * half;
            x1[row * CH + j] = xin[row * CH + j] + acc[t][v] + pb[j];
        }
    }
}

// ---------------------------------------------------------------------------
// GEMM4: u = gelu(m_bf @ fc1_w + fc1_b) -> bf16 (65536 x 768)
// ---------------------------------------------------------------------------
__global__ __launch_bounds__(256)
void gemm4_kernel(const bf16* __restrict__ A, const bf16* __restrict__ Bp,
                  const float* __restrict__ fb1, bf16* __restrict__ u, int mtiles) {
    const int NTW = 4, GROUPS = 12;           // 48 n-tiles
    const int wave = threadIdx.x >> 5, lane = threadIdx.x & 31;
    const int grp = blockIdx.x * 8 + wave;
    if (grp >= mtiles * GROUPS) return;
    const long m0  = (long)(grp / GROUPS) * 16;
    const int  nt0 = (grp % GROUPS) * NTW;
    v8f acc[NTW];
    wmma_strip<NTW>(A, CH, Bp, CH, m0, nt0, lane, acc);
    const int half = lane >> 4, l15 = lane & 15;
#pragma unroll
    for (int t = 0; t < NTW; ++t) {
        const int j = (nt0 + t) * 16 + l15;
#pragma unroll
        for (int v = 0; v < 8; ++v) {
            const long row = m0 + v + 8 * half;
            u[row * HDIM + j] = (bf16)gelu_f(acc[t][v] + fb1[j]);
        }
    }
}

// ---------------------------------------------------------------------------
// GEMM5: out = x1 + u_bf @ fc2_w + fc2_b  (K = 768)
// ---------------------------------------------------------------------------
__global__ __launch_bounds__(256)
void gemm5_kernel(const bf16* __restrict__ A, const bf16* __restrict__ Bp,
                  const float* __restrict__ fb2, const float* __restrict__ x1,
                  float* __restrict__ out, int mtiles) {
    const int NTW = 4, GROUPS = 3;
    const int wave = threadIdx.x >> 5, lane = threadIdx.x & 31;
    const int grp = blockIdx.x * 8 + wave;
    if (grp >= mtiles * GROUPS) return;
    const long m0  = (long)(grp / GROUPS) * 16;
    const int  nt0 = (grp % GROUPS) * NTW;
    v8f acc[NTW];
    wmma_strip<NTW>(A, HDIM, Bp, HDIM, m0, nt0, lane, acc);
    const int half = lane >> 4, l15 = lane & 15;
#pragma unroll
    for (int t = 0; t < NTW; ++t) {
        const int j = (nt0 + t) * 16 + l15;
#pragma unroll
        for (int v = 0; v < 8; ++v) {
            const long row = m0 + v + 8 * half;
            out[row * CH + j] = x1[row * CH + j] + acc[t][v] + fb2[j];
        }
    }
}

// ---------------------------------------------------------------------------
// Depthwise conv (SAME, zero-pad) + exact gelu + gated accumulate
// ---------------------------------------------------------------------------
__global__ __launch_bounds__(256)
void dwconv_kernel(const float* __restrict__ in, const float* __restrict__ kw,
                   const float* __restrict__ gates, int level, int ks,
                   float* __restrict__ outctx, float* __restrict__ ctx_all,
                   long total) {
    long i = (long)blockIdx.x * blockDim.x + threadIdx.x;
    if (i >= total) return;
    const int  c  = (int)(i % CH);
    const long sp = i / CH;
    const int  w  = (int)(sp & 63);
    const int  h  = (int)((sp >> 6) & 63);
    const long bb = sp & ~(SPATIAL - 1);
    const int  r  = ks >> 1;
    float sum = 0.f;
    for (int kh = 0; kh < ks; ++kh) {
        const int hh = h + kh - r;
        if ((unsigned)hh >= 64u) continue;
        for (int kwi = 0; kwi < ks; ++kwi) {
            const int ww = w + kwi - r;
            if ((unsigned)ww >= 64u) continue;
            sum += in[(bb + (long)hh * 64 + ww) * CH + c] *
                   kw[(kh * ks + kwi) * CH + c];
        }
    }
    const float g  = gelu_f(sum);
    outctx[i] = g;
    const float gt = gates[sp * 4 + level];
    if (level == 0) ctx_all[i] = g * gt;
    else            ctx_all[i] += g * gt;
}

// ---------------------------------------------------------------------------
// Spatial mean per (b,c) + gelu
// ---------------------------------------------------------------------------
__global__ __launch_bounds__(256)
void mean_gelu_kernel(const float* __restrict__ ctx, float* __restrict__ ctxg) {
    __shared__ float sm[256];
    const int b = blockIdx.x / CH, c = blockIdx.x % CH;
    float s = 0.f;
    for (int sp = threadIdx.x; sp < (int)SPATIAL; sp += 256)
        s += ctx[((long)b * SPATIAL + sp) * CH + c];
    sm[threadIdx.x] = s;
    __syncthreads();
    for (int st = 128; st > 0; st >>= 1) {
        if ((int)threadIdx.x < st) sm[threadIdx.x] += sm[threadIdx.x + st];
        __syncthreads();
    }
    if (threadIdx.x == 0) ctxg[b * CH + c] = gelu_f(sm[0] * (1.0f / SPATIAL));
}

// ---------------------------------------------------------------------------
// ctx_all += ctx_global * gate3 ; convert to bf16
// ---------------------------------------------------------------------------
__global__ __launch_bounds__(256)
void addglobal_kernel(const float* __restrict__ ctx_all, const float* __restrict__ ctxg,
                      const float* __restrict__ gates, bf16* __restrict__ outbf,
                      long total) {
    long i = (long)blockIdx.x * blockDim.x + threadIdx.x;
    if (i >= total) return;
    const int  c = (int)(i % CH);
    const long n = i / CH;
    const long b = n >> 12;
    outbf[i] = (bf16)(ctx_all[i] + ctxg[b * CH + c] * gates[n * 4 + 3]);
}

// ---------------------------------------------------------------------------
// Host-side launch orchestration
// ---------------------------------------------------------------------------
extern "C" void kernel_launch(void* const* d_in, const int* in_sizes, int n_in,
                              void* d_out, int out_size, void* d_ws, size_t ws_size,
                              hipStream_t stream) {
    (void)in_sizes; (void)n_in; (void)out_size; (void)ws_size;

    const float* x   = (const float*)d_in[0];
    const float* n1w = (const float*)d_in[1];
    const float* n1b = (const float*)d_in[2];
    const float* f_w = (const float*)d_in[3];
    const float* f_b = (const float*)d_in[4];
    const float* dw0 = (const float*)d_in[5];
    const float* dw1 = (const float*)d_in[6];
    const float* dw2 = (const float*)d_in[7];
    const float* h_w = (const float*)d_in[8];
    const float* h_b = (const float*)d_in[9];
    const float* p_w = (const float*)d_in[10];
    const float* p_b = (const float*)d_in[11];
    const float* n2w = (const float*)d_in[12];
    const float* n2b = (const float*)d_in[13];
    const float* f1w = (const float*)d_in[14];
    const float* f1b = (const float*)d_in[15];
    const float* f2w = (const float*)d_in[16];
    const float* f2b = (const float*)d_in[17];
    float* out = (float*)d_out;

    // Workspace layout (lifetime-aliased; packed weights same size as padded)
    char* ws = (char*)d_ws;
    bf16*  Wf      = (bf16*) (ws + 0);          // 400/16*192/32*512*2 = 153600
    bf16*  Wh      = (bf16*) (ws + 153600);     // 73728
    bf16*  Wp      = (bf16*) (ws + 227328);     // 73728
    bf16*  W1      = (bf16*) (ws + 301056);     // 294912
    bf16*  W2      = (bf16*) (ws + 595968);     // 294912
    float* ctxg    = (float*)(ws + 890880);     // 12288
    float* gates   = (float*)(ws + 903168);     // 1048576
    bf16*  bfA     = (bf16*) (ws + 1951744);    // 25165824 (y_bf / qmod_bf / m_bf)
    bf16*  bfB     = (bf16*) (ws + 27117568);   // 25165824 (ctx_all bf16)
    float* q       = (float*)(ws + 52283392);   // 50331648
    float* x1      = q;                         // alias: q dead after GEMM2
    float* ctxA    = (float*)(ws + 102615040);  // 50331648
    float* ctxB    = (float*)(ws + 152946688);  // 50331648
    float* ctx_all = (float*)(ws + 203278336);  // 50331648  (total ~254 MB)
    bf16*  u_bf    = (bf16*)ctxA;               // alias: 100 MB (ctxA+ctxB)

    const long TOT   = TOKENS * CH;
    const int  MT    = (int)(TOKENS / 16);      // 4096 M-tiles
    const int  EWBLK = (int)((TOT + 255) / 256);

    // 1) weight pack fp32 -> bf16 WMMA-B fragment layout
    wpack_kernel<<<(192 / 32 * (NFPAD / 16) * 512 + 255) / 256, 256, 0, stream>>>(
        f_w, Wf, 192, 388, NFPAD);
    wpack_kernel<<<(192 / 32 * 12 * 512 + 255) / 256, 256, 0, stream>>>(
        h_w, Wh, 192, 192, 192);
    wpack_kernel<<<(192 / 32 * 12 * 512 + 255) / 256, 256, 0, stream>>>(
        p_w, Wp, 192, 192, 192);
    wpack_kernel<<<(192 / 32 * 48 * 512 + 255) / 256, 256, 0, stream>>>(
        f1w, W1, 192, 768, 768);
    wpack_kernel<<<(768 / 32 * 12 * 512 + 255) / 256, 256, 0, stream>>>(
        f2w, W2, 768, 192, 192);

    // 2) LN1 -> bf16
    ln_bf16_kernel<<<(int)(TOKENS / 8), 256, 0, stream>>>(x, n1w, n1b, bfA, TOKENS);

    // 3) GEMM1: split q / ctx / gates   (5 groups of 5 tiles per M row)
    gemm1_kernel<<<MT * 5 / 8, 256, 0, stream>>>(bfA, Wf, f_b, q, ctxA, gates, MT);

    // 4) focal context chain
    dwconv_kernel<<<EWBLK, 256, 0, stream>>>(ctxA, dw0, gates, 0, 3, ctxB, ctx_all, TOT);
    dwconv_kernel<<<EWBLK, 256, 0, stream>>>(ctxB, dw1, gates, 1, 5, ctxA, ctx_all, TOT);
    dwconv_kernel<<<EWBLK, 256, 0, stream>>>(ctxA, dw2, gates, 2, 7, ctxB, ctx_all, TOT);

    // 5) global context
    mean_gelu_kernel<<<16 * CH, 256, 0, stream>>>(ctxB, ctxg);

    // 6) ctx_all += ctx_global * gate3 -> bf16
    addglobal_kernel<<<EWBLK, 256, 0, stream>>>(ctx_all, ctxg, gates, bfB, TOT);

    // 7) GEMM2: modulator fused with q*mod -> bf16
    gemm2_kernel<<<MT * 3 / 8, 256, 0, stream>>>(bfB, Wh, h_b, q, bfA, MT);

    // 8) GEMM3: proj + residual -> x1
    gemm3_kernel<<<MT * 3 / 8, 256, 0, stream>>>(bfA, Wp, p_b, x, x1, MT);

    // 9) LN2 -> bf16
    ln_bf16_kernel<<<(int)(TOKENS / 8), 256, 0, stream>>>(x1, n2w, n2b, bfA, TOKENS);

    // 10) GEMM4: fc1 + gelu -> bf16
    gemm4_kernel<<<MT * 12 / 8, 256, 0, stream>>>(bfA, W1, f1b, u_bf, MT);

    // 11) GEMM5: fc2 + residual -> out
    gemm5_kernel<<<MT * 3 / 8, 256, 0, stream>>>(u_bf, W2, f2b, x1, out, MT);
}
